// TransformerDecoder_33483565039931
// MI455X (gfx1250) — matrix operand order repaired and checked by reading, not compiled, not run on hardware
//
#include <hip/hip_runtime.h>

// ---------------------------------------------------------------------------
// GPT-2 style decoder forward for MI455X (gfx1250).
// All GEMMs: bf16 operands, v_wmma_f32_16x16x32_bf16, fp32 accumulate.
// Weights are pre-transposed to [N][K] bf16 so every LDS tile stage is a
// contiguous b128 copy -> async global->LDS path + double buffer.
// ---------------------------------------------------------------------------

typedef __bf16 bf16_t;
typedef __bf16 v8bf  __attribute__((ext_vector_type(8)));
typedef __bf16 v16bf __attribute__((ext_vector_type(16)));
typedef float  v8f   __attribute__((ext_vector_type(8)));
typedef int    v4i   __attribute__((ext_vector_type(4)));

#define WMMA_BF16(a, b, c) \
  __builtin_amdgcn_wmma_f32_16x16x32_bf16(false, (a), false, (b), (short)0, (c), false, false)

__device__ __forceinline__ v16bf cat8(v8bf lo, v8bf hi) {
  return __builtin_shufflevector(lo, hi, 0, 1, 2, 3, 4, 5, 6, 7,
                                 8, 9, 10, 11, 12, 13, 14, 15);
}

#if __has_builtin(__builtin_amdgcn_global_load_async_to_lds_b128)
#define HAVE_ASYNC_LDS 1
__device__ __forceinline__ void g2l_async16(const bf16_t* g, bf16_t* l) {
  __builtin_amdgcn_global_load_async_to_lds_b128(
      (__attribute__((address_space(1))) v4i*)(void*)(bf16_t*)g,
      (__attribute__((address_space(3))) v4i*)(void*)l, 0, 0);
}
__device__ __forceinline__ void wait_async0() {
#if __has_builtin(__builtin_amdgcn_s_wait_asynccnt)
  __builtin_amdgcn_s_wait_asynccnt(0);
#else
  asm volatile("s_wait_asynccnt 0" ::: "memory");
#endif
}
#endif

// Model constants
#define NB   4
#define NT   1024
#define ND   768
#define NH   12
#define NHD  64
#define NL   6
#define NF   3072
#define NV   50257
#define NM   4096        // B*T rows

// ---------------------------------------------------------------------------
// fp32 [K][N] -> bf16 [N][K] transposed weight converter (LDS tiled)
// blockIdx.z indexes the layer (offset K*N per layer).
// ---------------------------------------------------------------------------
__global__ __launch_bounds__(256) void k_transpose_bf16(const float* __restrict__ S,
                                                        bf16_t* __restrict__ D,
                                                        int K, int N) {
  __shared__ float t[32][33];
  size_t off = (size_t)blockIdx.z * K * N;
  const float* s = S + off;
  bf16_t* d = D + off;
  int kb = blockIdx.y * 32, nb = blockIdx.x * 32;
  int tx = threadIdx.x & 31, ty = threadIdx.x >> 5;
  for (int i = 0; i < 4; ++i) {
    int k = kb + ty + i * 8, n = nb + tx;
    t[ty + i * 8][tx] = (n < N) ? s[(size_t)k * N + n] : 0.f;
  }
  __syncthreads();
  for (int i = 0; i < 4; ++i) {
    int n = nb + ty + i * 8, k = kb + tx;
    if (n < N) d[(size_t)n * K + k] = (bf16_t)t[tx][ty + i * 8];
  }
}

// ---------------------------------------------------------------------------
// x = tok_emb[idx] + pos_emb[:T]
// ---------------------------------------------------------------------------
__global__ __launch_bounds__(256) void k_embed(const int* __restrict__ idx,
                                               const float* __restrict__ tok,
                                               const float* __restrict__ pos,
                                               float* __restrict__ x) {
  int row = blockIdx.x;
  int t   = row & (NT - 1);
  int tk  = idx[row];
  const float* tp = tok + (size_t)tk * ND;
  const float* pp = pos + (size_t)t * ND;
  float* xp = x + (size_t)row * ND;
  for (int i = threadIdx.x; i < ND; i += 256) xp[i] = tp[i] + pp[i];
}

// ---------------------------------------------------------------------------
// LayerNorm (block per row), bf16 output
// ---------------------------------------------------------------------------
__global__ __launch_bounds__(256) void k_layernorm(const float* __restrict__ X,
                                                   const float* __restrict__ S,
                                                   const float* __restrict__ Bp,
                                                   bf16_t* __restrict__ out) {
  __shared__ float sh[256];
  __shared__ float stats[2];
  int tid = threadIdx.x;
  int row = blockIdx.x;
  const float* x = X + (size_t)row * ND;
  float vals[3];
  float s = 0.f, sq = 0.f;
  for (int i = 0; i < 3; ++i) {
    float t = x[tid + i * 256];
    vals[i] = t;
    s += t;
    sq += t * t;
  }
  sh[tid] = s;
  __syncthreads();
  for (int o = 128; o > 0; o >>= 1) {
    if (tid < o) sh[tid] += sh[tid + o];
    __syncthreads();
  }
  if (tid == 0) stats[0] = sh[0] * (1.f / ND);
  __syncthreads();
  sh[tid] = sq;
  __syncthreads();
  for (int o = 128; o > 0; o >>= 1) {
    if (tid < o) sh[tid] += sh[tid + o];
    __syncthreads();
  }
  if (tid == 0) {
    float m = stats[0];
    stats[1] = rsqrtf(sh[0] * (1.f / ND) - m * m + 1e-5f);
  }
  __syncthreads();
  float m = stats[0], r = stats[1];
  bf16_t* op = out + (size_t)row * ND;
  for (int i = 0; i < 3; ++i) {
    int c = tid + i * 256;
    op[c] = (bf16_t)((vals[i] - m) * r * S[c] + Bp[c]);
  }
}

// ---------------------------------------------------------------------------
// Tiled bf16 WMMA GEMM: C[M,N] = A[M,K] * Wt[N,K]^T (+bias,+residual,ReLU)
// Block tile 128x64, BK=32, 8 waves -> each wave owns a 32x32 tile.
// Both LDS tiles staged as contiguous b128 copies; async + double buffered
// when the gfx1250 async-to-LDS builtin is available.
// ---------------------------------------------------------------------------
#define BM 128
#define BN 64
#define BK 32
#define KP 48     // padded LDS K stride (16B-aligned rows, bank offset)

__global__ __launch_bounds__(256) void k_gemm_bf16(
    const bf16_t* __restrict__ A, const bf16_t* __restrict__ Wt,
    const float* __restrict__ bias, const float* __restrict__ resid,
    float* __restrict__ outF, bf16_t* __restrict__ outB,
    int M, int N, int K, int doRelu) {
  __shared__ bf16_t As[2][BM][KP];
  __shared__ bf16_t Bs[2][BN][KP];
  int tid  = threadIdx.x;
  int lane = tid & 31, wv = tid >> 5;
  int half = lane >> 4, idx = lane & 15;
  int wm = (wv & 3) * 32;
  int wn = (wv >> 2) * 32;
  int m0 = blockIdx.y * BM, n0 = blockIdx.x * BN;

  // branch-free staging coordinates (one b128 chunk each)
  int arow0 = tid >> 2, akc = (tid & 3) * 8;   // A rows 0..63
  int arow1 = arow0 + 64;                      // A rows 64..127
  int brow  = tid >> 2;                        // B rows 0..63
  int bn = n0 + brow;
  if (bn > N - 1) bn = N - 1;                  // clamp; epilogue guards n<N
  const bf16_t* aptr0 = &A[(size_t)(m0 + arow0) * K + akc];
  const bf16_t* aptr1 = &A[(size_t)(m0 + arow1) * K + akc];
  const bf16_t* bptr  = &Wt[(size_t)bn * K + akc];

  v8f acc[2][2] = {};
  int nk = K / BK;

#if defined(HAVE_ASYNC_LDS)
  // prologue: stage tile 0
  g2l_async16(aptr0, &As[0][arow0][akc]);
  g2l_async16(aptr1, &As[0][arow1][akc]);
  g2l_async16(bptr,  &Bs[0][brow][akc]);
  for (int t = 0; t < nk; ++t) {
    wait_async0();
    __syncthreads();
    if (t + 1 < nk) {                          // stage next tile (alt buffer)
      int k0 = (t + 1) * BK, nb = (t + 1) & 1;
      g2l_async16(aptr0 + k0, &As[nb][arow0][akc]);
      g2l_async16(aptr1 + k0, &As[nb][arow1][akc]);
      g2l_async16(bptr + k0,  &Bs[nb][brow][akc]);
    }
    int b = t & 1;
#else
  for (int t = 0; t < nk; ++t) {
    int k0 = t * BK;
    __syncthreads();
    v8bf ra0 = *(const v8bf*)(aptr0 + k0);
    v8bf ra1 = *(const v8bf*)(aptr1 + k0);
    v8bf rb  = *(const v8bf*)(bptr + k0);
    *(v8bf*)&As[0][arow0][akc] = ra0;
    *(v8bf*)&As[0][arow1][akc] = ra1;
    *(v8bf*)&Bs[0][brow][akc]  = rb;
    __syncthreads();
    int b = 0;
#endif
    v16bf fa0, fa1, fb0, fb1;
    {
      int r = wm + idx;
      fa0 = cat8(*(const v8bf*)&As[b][r][half * 8],
                 *(const v8bf*)&As[b][r][16 + half * 8]);
      r = wm + 16 + idx;
      fa1 = cat8(*(const v8bf*)&As[b][r][half * 8],
                 *(const v8bf*)&As[b][r][16 + half * 8]);
      int nr = wn + idx;
      fb0 = cat8(*(const v8bf*)&Bs[b][nr][half * 8],
                 *(const v8bf*)&Bs[b][nr][16 + half * 8]);
      nr = wn + 16 + idx;
      fb1 = cat8(*(const v8bf*)&Bs[b][nr][half * 8],
                 *(const v8bf*)&Bs[b][nr][16 + half * 8]);
    }
    acc[0][0] = WMMA_BF16(fa0, fb0, acc[0][0]);
    acc[0][1] = WMMA_BF16(fa0, fb1, acc[0][1]);
    acc[1][0] = WMMA_BF16(fa1, fb0, acc[1][0]);
    acc[1][1] = WMMA_BF16(fa1, fb1, acc[1][1]);
  }

  // epilogue: C layout m = r + 8*half (per 16-lane group), n = idx
  for (int i = 0; i < 2; ++i)
    for (int j = 0; j < 2; ++j)
      for (int r = 0; r < 8; ++r) {
        int m = m0 + wm + i * 16 + r + half * 8;
        int n = n0 + wn + j * 16 + idx;
        if (n < N) {
          float v = acc[i][j][r];
          if (bias) v += bias[n];
          if (resid) v += resid[(size_t)m * N + n];
          if (doRelu) v = fmaxf(v, 0.f);
          if (outF) outF[(size_t)m * N + n] = v;
          if (outB) outB[(size_t)m * N + n] = (bf16_t)v;
        }
      }
}

// ---------------------------------------------------------------------------
// Flash-style causal attention. Block = 4 waves; wave owns 16 query rows.
// QK^T and P*V on WMMA; online softmax with 16-lane shfl reductions.
// ---------------------------------------------------------------------------
__global__ __launch_bounds__(128) void k_attention(const bf16_t* __restrict__ Q,
                                                   const bf16_t* __restrict__ Kb,
                                                   const bf16_t* __restrict__ Vb,
                                                   bf16_t* __restrict__ Ob,
                                                   float scale) {
  __shared__ bf16_t Ps[4][16][32];   // per-wave P tile (A layout staging)
  __shared__ bf16_t Vt[4][64][32];   // per-wave V^T tile: Vt[d][s]
  int tid  = threadIdx.x;
  int lane = tid & 31, wv = tid >> 5;
  int half = lane >> 4, idx = lane & 15;

  int blk  = blockIdx.x;             // B*H*(T/64)
  int tile = blk & 15;
  int h    = (blk >> 4) % NH;
  int b    = blk / (16 * NH);
  int q0   = tile * 64 + wv * 16;
  int rowb = b * NT;
  int hoff = h * NHD;

  v16bf aq[2];
  for (int dj = 0; dj < 2; ++dj) {
    const bf16_t* p = &Q[(size_t)(rowb + q0 + idx) * ND + hoff + dj * 32];
    aq[dj] = cat8(*(const v8bf*)(p + half * 8),
                  *(const v8bf*)(p + 16 + half * 8));
  }

  float mrow[8], lrow[8];
  v8f oacc[4] = {};
  for (int r = 0; r < 8; ++r) { mrow[r] = -1e30f; lrow[r] = 0.f; }

  for (int s0 = 0; s0 < q0 + 16; s0 += 32) {
    v8f sc[2];
    for (int sj = 0; sj < 2; ++sj) {
      v8f z = {};
      const bf16_t* kp = &Kb[(size_t)(rowb + s0 + sj * 16 + idx) * ND + hoff];
      for (int dj = 0; dj < 2; ++dj) {
        v16bf kf = cat8(*(const v8bf*)(kp + dj * 32 + half * 8),
                        *(const v8bf*)(kp + dj * 32 + 16 + half * 8));
        z = WMMA_BF16(aq[dj], kf, z);
      }
      sc[sj] = z;
    }
    for (int sj = 0; sj < 2; ++sj)
      for (int r = 0; r < 8; ++r) {
        int m = q0 + r + half * 8;
        int s = s0 + sj * 16 + idx;
        float v = sc[sj][r] * scale;
        sc[sj][r] = (s <= m) ? v : -1e30f;
      }
    float p0[8], p1[8];
    for (int r = 0; r < 8; ++r) {
      float mx = fmaxf(sc[0][r], sc[1][r]);
      for (int o = 1; o < 16; o <<= 1) mx = fmaxf(mx, __shfl_xor(mx, o, 32));
      float nm = fmaxf(mrow[r], mx);
      float corr = __expf(mrow[r] - nm);
      float e0 = __expf(sc[0][r] - nm);
      float e1 = __expf(sc[1][r] - nm);
      float rs = e0 + e1;
      for (int o = 1; o < 16; o <<= 1) rs += __shfl_xor(rs, o, 32);
      lrow[r] = lrow[r] * corr + rs;
      mrow[r] = nm;
      for (int dt = 0; dt < 4; ++dt) oacc[dt][r] *= corr;
      p0[r] = e0; p1[r] = e1;
    }
    for (int r = 0; r < 8; ++r) {
      int m = r + half * 8;
      Ps[wv][m][idx]      = (bf16_t)p0[r];
      Ps[wv][m][16 + idx] = (bf16_t)p1[r];
    }
    {
      const bf16_t* vp = &Vb[(size_t)(rowb + s0 + lane) * ND + hoff];
      v8bf vr[8];
      for (int c = 0; c < 8; ++c) vr[c] = *(const v8bf*)(vp + c * 8);
      for (int c = 0; c < 8; ++c)
        for (int e = 0; e < 8; ++e) Vt[wv][c * 8 + e][lane] = vr[c][e];
    }
    asm volatile("s_wait_dscnt 0" ::: "memory");  // wave-local LDS RAW
    v16bf pf = cat8(*(const v8bf*)&Ps[wv][idx][half * 8],
                    *(const v8bf*)&Ps[wv][idx][16 + half * 8]);
    for (int dt = 0; dt < 4; ++dt) {
      v16bf vf = cat8(*(const v8bf*)&Vt[wv][dt * 16 + idx][half * 8],
                      *(const v8bf*)&Vt[wv][dt * 16 + idx][16 + half * 8]);
      oacc[dt] = WMMA_BF16(pf, vf, oacc[dt]);
    }
  }

  for (int r = 0; r < 8; ++r) {
    float inv = 1.f / lrow[r];
    size_t row = (size_t)(rowb + q0 + r + half * 8);
    for (int dt = 0; dt < 4; ++dt)
      Ob[row * ND + hoff + dt * 16 + idx] = (bf16_t)(oacc[dt][r] * inv);
  }
}

// ---------------------------------------------------------------------------
// Loss: per-row log-softmax NLL, then mean reduce
// ---------------------------------------------------------------------------
__global__ __launch_bounds__(256) void k_loss_rows(const float* __restrict__ logits,
                                                   const int* __restrict__ tgt,
                                                   float* __restrict__ rows) {
  __shared__ float sh[256];
  int tid = threadIdx.x;
  int row = blockIdx.x;
  const float* lp = logits + (size_t)row * NV;
  float mx = -1e30f;
  for (int i = tid; i < NV; i += 256) mx = fmaxf(mx, lp[i]);
  sh[tid] = mx;
  __syncthreads();
  for (int o = 128; o > 0; o >>= 1) {
    if (tid < o) sh[tid] = fmaxf(sh[tid], sh[tid + o]);
    __syncthreads();
  }
  mx = sh[0];
  __syncthreads();
  float s = 0.f;
  for (int i = tid; i < NV; i += 256) s += __expf(lp[i] - mx);
  sh[tid] = s;
  __syncthreads();
  for (int o = 128; o > 0; o >>= 1) {
    if (tid < o) sh[tid] += sh[tid + o];
    __syncthreads();
  }
  if (tid == 0) rows[row] = -(lp[tgt[row]] - mx - __logf(sh[0]));
}

__global__ __launch_bounds__(256) void k_loss_reduce(const float* __restrict__ rows,
                                                     float* __restrict__ out) {
  __shared__ float sh[256];
  int tid = threadIdx.x;
  float s = 0.f;
  for (int i = tid; i < NM; i += 256) s += rows[i];
  sh[tid] = s;
  __syncthreads();
  for (int o = 128; o > 0; o >>= 1) {
    if (tid < o) sh[tid] += sh[tid + o];
    __syncthreads();
  }
  if (tid == 0) out[0] = sh[0] * (1.f / NM);
}

// ---------------------------------------------------------------------------
// Host orchestration
// ---------------------------------------------------------------------------
extern "C" void kernel_launch(void* const* d_in, const int* in_sizes, int n_in,
                              void* d_out, int out_size, void* d_ws, size_t ws_size,
                              hipStream_t stream) {
  const int*   idx     = (const int*)d_in[0];
  const int*   targets = (const int*)d_in[1];
  const float* tok     = (const float*)d_in[2];
  const float* pos     = (const float*)d_in[3];
  const float* ln1_s   = (const float*)d_in[4];
  const float* ln1_b   = (const float*)d_in[5];
  const float* wq      = (const float*)d_in[6];
  const float* wk      = (const float*)d_in[7];
  const float* wvv     = (const float*)d_in[8];
  const float* wo      = (const float*)d_in[9];
  const float* bo      = (const float*)d_in[10];
  const float* ln2_s   = (const float*)d_in[11];
  const float* ln2_b   = (const float*)d_in[12];
  const float* w1      = (const float*)d_in[13];
  const float* b1      = (const float*)d_in[14];
  const float* w2      = (const float*)d_in[15];
  const float* b2      = (const float*)d_in[16];
  const float* lnf_s   = (const float*)d_in[17];
  const float* lnf_b   = (const float*)d_in[18];
  const float* lm_w    = (const float*)d_in[19];
  const float* lm_b    = (const float*)d_in[20];

  char* p = (char*)d_ws;
  auto alloc = [&](size_t bytes) -> void* {
    void* r = (void*)p;
    p += (bytes + 255) & ~(size_t)255;
    return r;
  };
  const size_t DD = (size_t)ND * ND;
  const size_t DF = (size_t)ND * NF;
  bf16_t* wq_t  = (bf16_t*)alloc(NL * DD * 2);
  bf16_t* wk_t  = (bf16_t*)alloc(NL * DD * 2);
  bf16_t* wv_t  = (bf16_t*)alloc(NL * DD * 2);
  bf16_t* wo_t  = (bf16_t*)alloc(NL * DD * 2);
  bf16_t* w1_t  = (bf16_t*)alloc(NL * DF * 2);
  bf16_t* w2_t  = (bf16_t*)alloc(NL * DF * 2);
  bf16_t* lm_t  = (bf16_t*)alloc((size_t)NV * ND * 2);
  float*  x     = (float*)alloc((size_t)NM * ND * 4);
  bf16_t* h_bf  = (bf16_t*)alloc((size_t)NM * ND * 2);
  bf16_t* q_bf  = (bf16_t*)alloc((size_t)NM * ND * 2);
  bf16_t* k_bf  = (bf16_t*)alloc((size_t)NM * ND * 2);
  bf16_t* v_bf  = (bf16_t*)alloc((size_t)NM * ND * 2);
  bf16_t* att_bf= (bf16_t*)alloc((size_t)NM * ND * 2);
  bf16_t* mlp_bf= (bf16_t*)alloc((size_t)NM * NF * 2);
  float*  rowls = (float*)alloc((size_t)NM * 4);

  // 1) weight conversion + transpose to [N][K] bf16
  {
    dim3 gDD(24, 24, NL);                 // K=768, N=768
    k_transpose_bf16<<<gDD, 256, 0, stream>>>(wq,  wq_t, ND, ND);
    k_transpose_bf16<<<gDD, 256, 0, stream>>>(wk,  wk_t, ND, ND);
    k_transpose_bf16<<<gDD, 256, 0, stream>>>(wvv, wv_t, ND, ND);
    k_transpose_bf16<<<gDD, 256, 0, stream>>>(wo,  wo_t, ND, ND);
    dim3 g1(96, 24, NL);                  // K=768, N=3072
    k_transpose_bf16<<<g1, 256, 0, stream>>>(w1, w1_t, ND, NF);
    dim3 g2(24, 96, NL);                  // K=3072, N=768
    k_transpose_bf16<<<g2, 256, 0, stream>>>(w2, w2_t, NF, ND);
    dim3 gLM((NV + 31) / 32, 24, 1);      // K=768, N=50257
    k_transpose_bf16<<<gLM, 256, 0, stream>>>(lm_w, lm_t, ND, NV);
  }

  // 2) embeddings
  k_embed<<<NM, 256, 0, stream>>>(idx, tok, pos, x);

  const float scale = 0.03608439182435161f;  // 1/sqrt(768) (ref scales by D)
  dim3 gD(ND / BN, NM / BM);                 // 12 x 32
  dim3 gF(NF / BN, NM / BM);                 // 48 x 32
  dim3 gV((NV + BN - 1) / BN, NM / BM);      // 786 x 32

  // 3) transformer blocks
  for (int l = 0; l < NL; ++l) {
    k_layernorm<<<NM, 256, 0, stream>>>(x, ln1_s + l * ND, ln1_b + l * ND, h_bf);
    k_gemm_bf16<<<gD, 256, 0, stream>>>(h_bf, wq_t + l * DD, nullptr, nullptr,
                                        nullptr, q_bf, NM, ND, ND, 0);
    k_gemm_bf16<<<gD, 256, 0, stream>>>(h_bf, wk_t + l * DD, nullptr, nullptr,
                                        nullptr, k_bf, NM, ND, ND, 0);
    k_gemm_bf16<<<gD, 256, 0, stream>>>(h_bf, wv_t + l * DD, nullptr, nullptr,
                                        nullptr, v_bf, NM, ND, ND, 0);
    k_attention<<<NB * NH * (NT / 64), 128, 0, stream>>>(q_bf, k_bf, v_bf,
                                                         att_bf, scale);
    k_gemm_bf16<<<gD, 256, 0, stream>>>(att_bf, wo_t + l * DD, bo + l * ND, x,
                                        x, nullptr, NM, ND, ND, 0);
    k_layernorm<<<NM, 256, 0, stream>>>(x, ln2_s + l * ND, ln2_b + l * ND, h_bf);
    k_gemm_bf16<<<gF, 256, 0, stream>>>(h_bf, w1_t + l * DF, b1 + l * NF,
                                        nullptr, nullptr, mlp_bf, NM, NF, ND, 1);
    k_gemm_bf16<<<gD, 256, 0, stream>>>(mlp_bf, w2_t + l * DF, b2 + l * ND, x,
                                        x, nullptr, NM, ND, NF, 0);
  }

  // 4) final LN + LM head straight into d_out
  k_layernorm<<<NM, 256, 0, stream>>>(x, lnf_s, lnf_b, h_bf);
  float* logits = (float*)d_out;
  k_gemm_bf16<<<gV, 256, 0, stream>>>(h_bf, lm_t, lm_b, nullptr, logits,
                                      nullptr, NM, NV, ND, 0);

  // 5) loss
  k_loss_rows<<<NM, 256, 0, stream>>>(logits, targets, rowls);
  k_loss_reduce<<<1, 256, 0, stream>>>(rowls, logits + (size_t)NM * NV);
}
